// MultiheadSelfAttention_68401649156618
// MI455X (gfx1250) — compile-verified, matrix operand order checked by reading, not run on hardware
//
#include <hip/hip_runtime.h>
#include <hip/hip_bf16.h>

typedef __attribute__((ext_vector_type(16))) _Float16 v16h;
typedef __attribute__((ext_vector_type(4)))  _Float16 v4h;
typedef __attribute__((ext_vector_type(8)))  float    v8f;
typedef __attribute__((ext_vector_type(4)))  int      v4i;

#define BB   2
#define SEQ  2048
#define DM   1024
#define NH   16
#define HD   64          // head dim
#define MROWS (BB*SEQ)   // 4096

static __device__ __forceinline__ v8f wmma_f16(v16h a, v16h b, v8f c) {
    // D = A(16x32 f16) * B(32x16 f16) + C(16x16 f32)
    return __builtin_amdgcn_wmma_f32_16x16x32_f16(
        false, a, false, b, (short)0, c, false, false);
}

// ---------------------------------------------------------------------------
// 16-byte global -> LDS staging element. CDNA5 async LDS-DMA when available
// (probe confirmed the builtin exists; param type is int4* per the diagnostic).
// ---------------------------------------------------------------------------
#if __has_builtin(__builtin_amdgcn_global_load_async_to_lds_b128)
#define HAVE_ASYNC_LDS 1
#else
#define HAVE_ASYNC_LDS 0
#endif

static __device__ __forceinline__ void stage16(const _Float16* g, _Float16* l) {
#if HAVE_ASYNC_LDS
    __builtin_amdgcn_global_load_async_to_lds_b128(
        (__attribute__((address_space(1))) v4i*)g,
        (__attribute__((address_space(3))) v4i*)l, 0, 0);
#else
    *(uint4*)l = *(const uint4*)g;
#endif
}

static __device__ __forceinline__ void stage_wait() {
#if HAVE_ASYNC_LDS
    asm volatile("s_wait_asynccnt 0x0" ::: "memory");
#endif
}

// ---------------------------------------------------------------------------
// 1) f32 -> f16 downcast (x4 vectorized; all sizes are multiples of 4)
// ---------------------------------------------------------------------------
__global__ void cvt_f32_f16(const float* __restrict__ in,
                            _Float16* __restrict__ out, int n) {
    int i = (blockIdx.x * blockDim.x + threadIdx.x) * 4;
    if (i < n) {
        float4 v = *(const float4*)(in + i);
        v4h h = { (_Float16)v.x, (_Float16)v.y, (_Float16)v.z, (_Float16)v.w };
        *(v4h*)(out + i) = h;
    }
}

// ---------------------------------------------------------------------------
// 2) GEMM: C[M][N] = A[M][K] * W[N][K]^T, LDS double-buffered.
//    Block (256 thr, 8 waves) -> 128x128 tile, K-chunks of 64.
//    Wave -> 32x64 (8 f32 accumulators), fragments from LDS.
//    grid = (N/128, M/128)
// ---------------------------------------------------------------------------
#define LDSTR 80   // padded LDS row stride in halfs (32B aligned, bank-spread)

__global__ __launch_bounds__(256) void gemm_xwT_wmma(
    const _Float16* __restrict__ A,   // [M][K]
    const _Float16* __restrict__ Bw,  // [N][K]
    float* __restrict__ C,            // [M][N]
    int M, int N, int K)
{
    __shared__ __align__(32) _Float16 Al[2][128][LDSTR];
    __shared__ __align__(32) _Float16 Bl[2][128][LDSTR];

    const int t      = threadIdx.x;
    const int lane   = t & 31;
    const int w      = t >> 5;
    const int wm     = w >> 1;            // 0..3 : 32-row band
    const int wn     = w & 1;             // 0..1 : 64-col band
    const int rlo    = lane & 15;
    const int khalf  = (lane >> 4) * 16;
    const int roff   = (lane >> 4) * 8;

    const int mbase  = blockIdx.y * 128;
    const int nbase  = blockIdx.x * 128;

    // staging: thread -> one 64B (32-half) chunk of A and of B per K-chunk
    const int srow   = t >> 1;            // 0..127
    const int schunk = (t & 1) * 32;      // half offset within 64-wide chunk
    const _Float16* ag = A  + (size_t)(mbase + srow) * K + schunk;
    const _Float16* bg = Bw + (size_t)(nbase + srow) * K + schunk;

    v8f acc[2][4] = {};

    const int NC = K >> 6;   // chunks of 64

    // prologue: stage chunk 0 into buffer 0
#pragma unroll
    for (int i = 0; i < 4; ++i) {
        stage16(ag + i * 8, &Al[0][srow][schunk] + i * 8);
        stage16(bg + i * 8, &Bl[0][srow][schunk] + i * 8);
    }
    stage_wait();
    __syncthreads();

    int buf = 0;
    for (int c = 0; c < NC; ++c) {
        // prefetch chunk c+1 into the other buffer while computing chunk c
        if (c + 1 < NC) {
            const int k0 = (c + 1) * 64;
#pragma unroll
            for (int i = 0; i < 4; ++i) {
                stage16(ag + k0 + i * 8, &Al[buf ^ 1][srow][schunk] + i * 8);
                stage16(bg + k0 + i * 8, &Bl[buf ^ 1][srow][schunk] + i * 8);
            }
        }

        // compute chunk c from LDS: 2 k-steps x 8 WMMA
#pragma unroll
        for (int ks = 0; ks < 2; ++ks) {
            v16h a0 = *(const v16h*)&Al[buf][wm * 32 +      rlo][ks * 32 + khalf];
            v16h a1 = *(const v16h*)&Al[buf][wm * 32 + 16 + rlo][ks * 32 + khalf];
            v16h b0 = *(const v16h*)&Bl[buf][wn * 64 +      rlo][ks * 32 + khalf];
            v16h b1 = *(const v16h*)&Bl[buf][wn * 64 + 16 + rlo][ks * 32 + khalf];
            v16h b2 = *(const v16h*)&Bl[buf][wn * 64 + 32 + rlo][ks * 32 + khalf];
            v16h b3 = *(const v16h*)&Bl[buf][wn * 64 + 48 + rlo][ks * 32 + khalf];
            acc[0][0] = wmma_f16(a0, b0, acc[0][0]);
            acc[0][1] = wmma_f16(a0, b1, acc[0][1]);
            acc[0][2] = wmma_f16(a0, b2, acc[0][2]);
            acc[0][3] = wmma_f16(a0, b3, acc[0][3]);
            acc[1][0] = wmma_f16(a1, b0, acc[1][0]);
            acc[1][1] = wmma_f16(a1, b1, acc[1][1]);
            acc[1][2] = wmma_f16(a1, b2, acc[1][2]);
            acc[1][3] = wmma_f16(a1, b3, acc[1][3]);
        }

        stage_wait();      // async copies into buf^1 complete (no-op in fallback)
        __syncthreads();   // everyone done reading buf / writing buf^1
        buf ^= 1;
    }

    // epilogue: C layout rows r (lanes 0-15) / r+8 (lanes 16-31)
#pragma unroll
    for (int ms = 0; ms < 2; ++ms) {
        const int rowb = mbase + wm * 32 + ms * 16 + roff;
        float* crow = C + (size_t)rowb * N + nbase + wn * 64 + rlo;
#pragma unroll
        for (int r = 0; r < 8; ++r) {
            crow[(size_t)r * N +  0] = acc[ms][0][r];
            crow[(size_t)r * N + 16] = acc[ms][1][r];
            crow[(size_t)r * N + 32] = acc[ms][2][r];
            crow[(size_t)r * N + 48] = acc[ms][3][r];
        }
    }
    (void)M;
}

// ---------------------------------------------------------------------------
// 3) RoPE on Q,K + pack f16 [b,h,s,64]; V transposed [b,h,64,s]
// ---------------------------------------------------------------------------
__global__ void rope_pack(const float* __restrict__ Q, const float* __restrict__ Kq,
                          const float* __restrict__ V, const int* __restrict__ pos,
                          _Float16* __restrict__ Qh, _Float16* __restrict__ Kh,
                          _Float16* __restrict__ Vt)
{
    int i = blockIdx.x * blockDim.x + threadIdx.x;
    if (i >= BB * SEQ * NH * 32) return;
    int p = i & 31;
    int h = (i >> 5) & (NH - 1);
    int s = (i >> 9) & (SEQ - 1);
    int b = i >> 20;

    size_t src = ((size_t)(b * SEQ + s)) * DM + h * HD + 2 * p;
    float qe = Q[src],  qo = Q[src + 1];
    float ke = Kq[src], ko = Kq[src + 1];
    float ve = V[src],  vo = V[src + 1];

    float ang = (float)pos[s] * __powf(10000.0f, -(2.0f * (float)p) * (1.0f / 64.0f));
    float sn, cs;
    __sincosf(ang, &sn, &cs);

    size_t bh = (size_t)(b * NH + h);
    size_t dq = (bh * SEQ + s) * HD + 2 * p;
    Qh[dq]     = (_Float16)(cs * qe - sn * qo);
    Qh[dq + 1] = (_Float16)(sn * qe + cs * qo);
    Kh[dq]     = (_Float16)(cs * ke - sn * ko);
    Kh[dq + 1] = (_Float16)(sn * ke + cs * ko);

    size_t dv = (bh * HD + 2 * p) * SEQ + s;
    Vt[dv]       = (_Float16)ve;
    Vt[dv + SEQ] = (_Float16)vo;
}

// ---------------------------------------------------------------------------
// 4) Flash attention: one wave per (b,h, 32-query tile). Causal, online softmax.
//    Only the diagonal 32x32 block needs masking. K frags for kb+1 prefetched
//    during softmax; V frags issued before softmax.
// ---------------------------------------------------------------------------
static __device__ __forceinline__ void load_kfrags(v16h (&bk)[2][2],
                                                   const _Float16* base, int k0,
                                                   int rlo, int khalf) {
    const _Float16* kp = base + (size_t)(k0 + rlo) * HD + khalf;
    bk[0][0] = *(const v16h*)(kp);
    bk[0][1] = *(const v16h*)(kp + 32);
    bk[1][0] = *(const v16h*)(kp + (size_t)16 * HD);
    bk[1][1] = *(const v16h*)(kp + (size_t)16 * HD + 32);
}

__global__ __launch_bounds__(256) void flash_attn_wmma(
    const _Float16* __restrict__ Qh, const _Float16* __restrict__ Kh,
    const _Float16* __restrict__ Vt, _Float16* __restrict__ Ao)
{
    __shared__ __align__(64) _Float16 plds[8][32][32];

    const int lane  = threadIdx.x & 31;
    const int w     = threadIdx.x >> 5;
    const int gid   = blockIdx.x * 8 + w;
    const int qt    = gid & (SEQ / 32 - 1);   // 64 q-tiles of 32
    const int bh    = gid >> 6;
    const int b     = bh >> 4, h = bh & 15;
    const int qbase = qt * 32;
    const int rlo   = lane & 15;
    const int khalf = (lane >> 4) * 16;
    const int qoff  = (lane >> 4) * 8;

    // Q fragments: 2 q-subtiles x 2 k-halves
    v16h a[2][2];
#pragma unroll
    for (int qs = 0; qs < 2; ++qs) {
        const _Float16* qp = Qh + ((size_t)bh * SEQ + qbase + qs * 16 + rlo) * HD + khalf;
        a[qs][0] = *(const v16h*)(qp);
        a[qs][1] = *(const v16h*)(qp + 32);
    }

    v8f o[2][4] = {};
    float mst[2][8], lst[2][8];
#pragma unroll
    for (int qs = 0; qs < 2; ++qs)
#pragma unroll
        for (int r = 0; r < 8; ++r) { mst[qs][r] = -1e30f; lst[qs][r] = 0.0f; }

    const float scale = 0.125f;   // 1/sqrt(64)
    const int nkb = qt + 1;       // causal: blocks 0..qt; only kb==qt is diagonal

    const _Float16* kmat = Kh + (size_t)bh * SEQ * HD;
    v16h bk[2][2];
    load_kfrags(bk, kmat, 0, rlo, khalf);

    for (int kb = 0; kb < nkb; ++kb) {
        const int k0 = kb * 32;

        // V fragments (independent of scores -> in flight during softmax)
        const _Float16* vp = Vt + ((size_t)bh * HD + rlo) * SEQ + k0 + khalf;
        v16h bv0 = *(const v16h*)(vp);
        v16h bv1 = *(const v16h*)(vp + (size_t)16 * SEQ);
        v16h bv2 = *(const v16h*)(vp + (size_t)32 * SEQ);
        v16h bv3 = *(const v16h*)(vp + (size_t)48 * SEQ);

        // scores: 2 q-subtiles x 2 key-subtiles
        v8f s[2][2];
#pragma unroll
        for (int qs = 0; qs < 2; ++qs)
#pragma unroll
            for (int kt = 0; kt < 2; ++kt) {
                v8f z = {};
                z = wmma_f16(a[qs][0], bk[kt][0], z);
                z = wmma_f16(a[qs][1], bk[kt][1], z);
                s[qs][kt] = z;
            }

        // prefetch next K block while we do the softmax below
        if (kb + 1 < nkb) load_kfrags(bk, kmat, k0 + 32, rlo, khalf);

        const bool diag = (kb == qt);
        const int kc0 = k0 + rlo, kc1 = k0 + 16 + rlo;
#pragma unroll
        for (int qs = 0; qs < 2; ++qs) {
#pragma unroll
            for (int r = 0; r < 8; ++r) {
                const int q = qbase + qs * 16 + qoff + r;
                float v0 = s[qs][0][r] * scale;
                float v1 = s[qs][1][r] * scale;
                if (diag) {
                    if (kc0 > q) v0 = -1e30f;
                    if (kc1 > q) v1 = -1e30f;
                }
                float mx = fmaxf(v0, v1);
                mx = fmaxf(mx, __shfl_xor(mx, 1, 32));
                mx = fmaxf(mx, __shfl_xor(mx, 2, 32));
                mx = fmaxf(mx, __shfl_xor(mx, 4, 32));
                mx = fmaxf(mx, __shfl_xor(mx, 8, 32));
                float mnew = fmaxf(mst[qs][r], mx);
                float corr = __expf(mst[qs][r] - mnew);
                float p0 = __expf(v0 - mnew);
                float p1 = __expf(v1 - mnew);
                float rs = p0 + p1;
                rs += __shfl_xor(rs, 1, 32);
                rs += __shfl_xor(rs, 2, 32);
                rs += __shfl_xor(rs, 4, 32);
                rs += __shfl_xor(rs, 8, 32);
                lst[qs][r] = lst[qs][r] * corr + rs;
                mst[qs][r] = mnew;
                o[qs][0][r] *= corr; o[qs][1][r] *= corr;
                o[qs][2][r] *= corr; o[qs][3][r] *= corr;
                plds[w][qs * 16 + qoff + r][rlo]      = (_Float16)p0;
                plds[w][qs * 16 + qoff + r][16 + rlo] = (_Float16)p1;
            }
        }
        __threadfence_block();  // order wave-private LDS writes vs cross-lane reads

        v16h pa0 = *(const v16h*)(&plds[w][rlo][khalf]);
        v16h pa1 = *(const v16h*)(&plds[w][16 + rlo][khalf]);
        o[0][0] = wmma_f16(pa0, bv0, o[0][0]);
        o[0][1] = wmma_f16(pa0, bv1, o[0][1]);
        o[0][2] = wmma_f16(pa0, bv2, o[0][2]);
        o[0][3] = wmma_f16(pa0, bv3, o[0][3]);
        o[1][0] = wmma_f16(pa1, bv0, o[1][0]);
        o[1][1] = wmma_f16(pa1, bv1, o[1][1]);
        o[1][2] = wmma_f16(pa1, bv2, o[1][2]);
        o[1][3] = wmma_f16(pa1, bv3, o[1][3]);
    }

    // normalize + store f16 attention output [b, s, h*64 + d]
#pragma unroll
    for (int qs = 0; qs < 2; ++qs) {
        _Float16* aout = Ao + ((size_t)(b * SEQ) + qbase + qs * 16 + qoff) * DM
                            + h * HD + rlo;
#pragma unroll
        for (int r = 0; r < 8; ++r) {
            float inv = 1.0f / lst[qs][r];
            _Float16* orow = aout + (size_t)r * DM;
            orow[0]  = (_Float16)(o[qs][0][r] * inv);
            orow[16] = (_Float16)(o[qs][1][r] * inv);
            orow[32] = (_Float16)(o[qs][2][r] * inv);
            orow[48] = (_Float16)(o[qs][3][r] * inv);
        }
    }
}

// ---------------------------------------------------------------------------
// Launcher
// ---------------------------------------------------------------------------
extern "C" void kernel_launch(void* const* d_in, const int* in_sizes, int n_in,
                              void* d_out, int out_size, void* d_ws, size_t ws_size,
                              hipStream_t stream) {
    const float* x  = (const float*)d_in[0];
    const float* Wq = (const float*)d_in[1];
    const float* Wk = (const float*)d_in[2];
    const float* Wv = (const float*)d_in[3];
    const float* Wo = (const float*)d_in[4];
    const int*  pos = (const int*)d_in[5];
    float* out = (float*)d_out;

    // Workspace layout (~88 MB; attn output aliases dead Qf buffer)
    char* ws = (char*)d_ws;
    _Float16* xh  = (_Float16*)ws; ws += (size_t)MROWS * DM * 2;
    _Float16* Wqh = (_Float16*)ws; ws += (size_t)DM * DM * 2;
    _Float16* Wkh = (_Float16*)ws; ws += (size_t)DM * DM * 2;
    _Float16* Wvh = (_Float16*)ws; ws += (size_t)DM * DM * 2;
    _Float16* Woh = (_Float16*)ws; ws += (size_t)DM * DM * 2;
    float*    Qf  = (float*)ws;    ws += (size_t)MROWS * DM * 4;
    float*    Kf  = (float*)ws;    ws += (size_t)MROWS * DM * 4;
    float*    Vf  = (float*)ws;    ws += (size_t)MROWS * DM * 4;
    _Float16* Qh  = (_Float16*)ws; ws += (size_t)MROWS * DM * 2;
    _Float16* Kh  = (_Float16*)ws; ws += (size_t)MROWS * DM * 2;
    _Float16* Vt  = (_Float16*)ws; ws += (size_t)MROWS * DM * 2;
    _Float16* Ao  = (_Float16*)Qf;   // reuse: Qf dead after rope_pack

    // 1) downcast
    {
        int n = MROWS * DM;
        cvt_f32_f16<<<(n / 4 + 255) / 256, 256, 0, stream>>>(x, xh, n);
        int nw = DM * DM;
        cvt_f32_f16<<<(nw / 4 + 255) / 256, 256, 0, stream>>>(Wq, Wqh, nw);
        cvt_f32_f16<<<(nw / 4 + 255) / 256, 256, 0, stream>>>(Wk, Wkh, nw);
        cvt_f32_f16<<<(nw / 4 + 255) / 256, 256, 0, stream>>>(Wv, Wvh, nw);
        cvt_f32_f16<<<(nw / 4 + 255) / 256, 256, 0, stream>>>(Wo, Woh, nw);
    }

    // 2) QKV projections
    dim3 gg(DM / 128, MROWS / 128);   // (8, 32)
    gemm_xwT_wmma<<<gg, 256, 0, stream>>>(xh, Wqh, Qf, MROWS, DM, DM);
    gemm_xwT_wmma<<<gg, 256, 0, stream>>>(xh, Wkh, Kf, MROWS, DM, DM);
    gemm_xwT_wmma<<<gg, 256, 0, stream>>>(xh, Wvh, Vf, MROWS, DM, DM);

    // 3) RoPE + head-split pack
    {
        int n = BB * SEQ * NH * 32;
        rope_pack<<<n / 256, 256, 0, stream>>>(Qf, Kf, Vf, pos, Qh, Kh, Vt);
    }

    // 4) causal flash attention
    {
        int nwaves = BB * NH * (SEQ / 32);   // 2048
        flash_attn_wmma<<<nwaves / 8, 256, 0, stream>>>(Qh, Kh, Vt, Ao);
    }

    // 5) output projection -> f32 d_out
    gemm_xwT_wmma<<<gg, 256, 0, stream>>>(Ao, Woh, out, MROWS, DM, DM);
}
